// Qwen3Attention_57569741636391
// MI455X (gfx1250) — matrix operand order repaired and checked by reading, not compiled, not run on hardware
//
#include <hip/hip_runtime.h>

// ---------------------------------------------------------------------------
// Qwen3 attention block for MI455X (gfx1250, wave32, WMMA)
// B=1, S=2048, HID=2048, H=16, KH=8, D=128, GROUPS=2
// Heavy contractions: v_wmma_f32_16x16x32_f16, 64x64 per-wave tiles,
// LDS-staged B tiles via global_load_async_to_lds_b128 (ASYNCcnt path).
// ---------------------------------------------------------------------------

typedef __attribute__((ext_vector_type(16))) _Float16 v16h;
typedef __attribute__((ext_vector_type(8)))  _Float16 v8h;
typedef __attribute__((ext_vector_type(8)))  float    v8f;
typedef int v4i_ __attribute__((vector_size(16)));   // builtin's b128 unit

#define SQ    2048      // sequence length
#define HIDQ  2048      // hidden size
#define HQ    16        // q heads
#define KHQ   8         // kv heads
#define DQ    128       // head dim
#define GRPQ  (HQ / KHQ)
#define EPSQ  1e-6f
#define SCALEQ 0.08838834764831845f   // 1/sqrt(128)

#define BN    128       // block tile N
#define BM    128       // block tile M
#define BK    32        // k-step
#define BROW  40        // padded LDS row stride (halves) = 80B: conflict-free, 16B aligned

#if defined(__has_builtin)
#if __has_builtin(__builtin_amdgcn_global_load_async_to_lds_b128)
#define HAVE_ASYNC_LDS 1
#endif
#endif

typedef __attribute__((address_space(1))) v4i_* as1_v4i_p;
typedef __attribute__((address_space(3))) v4i_* as3_v4i_p;

__device__ __forceinline__ void async_copy_b128(const _Float16* g, _Float16* l) {
#ifdef HAVE_ASYNC_LDS
  __builtin_amdgcn_global_load_async_to_lds_b128((as1_v4i_p)g, (as3_v4i_p)l, 0, 0);
#else
  *(v8h*)l = *(const v8h*)g;
#endif
}

__device__ __forceinline__ void wait_async_lds() {
#ifdef HAVE_ASYNC_LDS
#if __has_builtin(__builtin_amdgcn_s_wait_asynccnt)
  __builtin_amdgcn_s_wait_asynccnt(0);
#else
  asm volatile("s_wait_asynccnt 0x0" ::: "memory");
#endif
#endif
}

// ---------------------------------------------------------------------------
// WMMA fragment helpers (layouts per CDNA5 ISA 7.12.2, 16-bit A 16x32 / B 32x16)
// ---------------------------------------------------------------------------

__device__ __forceinline__ v8f wmma_f16(v16h a, v16h b, v8f c) {
  return __builtin_amdgcn_wmma_f32_16x16x32_f16(false, a, false, b, (short)0, c,
                                                false, false);
}

// A: row-major M x K. Fragment rows [m0,m0+16), K [k0,k0+32).
__device__ __forceinline__ v16h load_frag_a(const _Float16* __restrict__ A,
                                            int lda, int m0, int k0, int lane) {
  int r = lane & 15;
  int h = lane >> 4;
  const _Float16* p = A + (size_t)(m0 + r) * lda + k0 + h * 8;
  union { v16h v; v8h h2[2]; } u;
  u.h2[0] = *(const v8h*)(p);
  u.h2[1] = *(const v8h*)(p + 16);
  return u.v;
}

// B fragment from the LDS-staged tile (tile row n = logical column, padded stride).
__device__ __forceinline__ v16h load_frag_b_lds(const _Float16* Bl, int n0, int lane) {
  int n = n0 + (lane & 15);
  int h = lane >> 4;
  const _Float16* p = Bl + n * BROW + h * 8;
  union { v16h v; v8h h2[2]; } u;
  u.h2[0] = *(const v8h*)(p);
  u.h2[1] = *(const v8h*)(p + 16);
  return u.v;
}

// C/D 16x16 f32: lane<16 -> N=lane, M=vgpr; lane>=16 -> N=lane-16, M=vgpr+8
__device__ __forceinline__ void store_frag_c(float* __restrict__ C, int ldc,
                                             int m0, int n0, v8f acc, int lane) {
  int n  = lane & 15;
  int mh = (lane >> 4) * 8;
#pragma unroll
  for (int r = 0; r < 8; ++r) {
    C[(size_t)(m0 + mh + r) * ldc + n0 + n] = acc[r];
  }
}

// Cooperative stage of one BN x BK f16 B tile into LDS (thread t -> tile row t).
__device__ __forceinline__ void stage_b_tile(const _Float16* __restrict__ B, int ldb,
                                             int nB, int k0, _Float16* dst, int tid) {
  const _Float16* g = B + (size_t)(nB + tid) * ldb + k0;
  _Float16* l = dst + tid * BROW;
#pragma unroll
  for (int c = 0; c < 4; ++c) async_copy_b128(g + c * 8, l + c * 8);
}

// ---------------------------------------------------------------------------
// GEMM: C(MxN, f32) = A(MxK, f16, row-major) * B^T  (B stored NxK f16)
// mode 0: plain   mode 1: causal block-skip (scores)  mode 2: causal K-limit (P*V)
// Block = 32x4 threads (4 waves, 2x2), wave tile 64x64 (16 WMMA / k-step),
// block tile 128x128, B tile double-buffered through LDS.
// M, N must be multiples of 128; K multiple of 32 (true for this problem).
// ---------------------------------------------------------------------------
__global__ __launch_bounds__(128)
void gemm_f16f32(const _Float16* __restrict__ A, const _Float16* __restrict__ B,
                 float* __restrict__ C, int M, int N, int K,
                 int lda, int ldb, int ldc, int mode) {
  __shared__ __align__(16) _Float16 Blds[2][BN * BROW];

  const int lane = threadIdx.x;                 // 0..31
  const int wv   = threadIdx.y;                 // 0..3
  const int tid  = wv * 32 + lane;              // 0..127
  const int mB   = blockIdx.y * BM;
  const int nB   = blockIdx.x * BN;
  if (mB >= M || nB >= N) return;

  if (mode == 1 && nB > mB + (BM - 1)) return;  // block entirely above diagonal

  int Keff = K;
  if (mode == 2) { int lim = mB + BM; Keff = (lim < K) ? lim : K; }

  const int m0 = mB + (wv >> 1) * 64;           // wave row origin (global M)
  const int nl = (wv & 1) * 64;                 // wave col origin (local to block)

  v8f acc[4][4] = {};

  stage_b_tile(B, ldb, nB, 0, Blds[0], tid);
  wait_async_lds();
  __syncthreads();

  int buf = 0;
  for (int k0 = 0; k0 < Keff; k0 += BK, buf ^= 1) {
    if (k0 + BK < Keff) stage_b_tile(B, ldb, nB, k0 + BK, Blds[buf ^ 1], tid);

    v16h a[4], b[4];
#pragma unroll
    for (int i = 0; i < 4; ++i) a[i] = load_frag_a(A, lda, m0 + 16 * i, k0, lane);
    if (k0 + 2 * BK < Keff) {
      // hint: next-next A fragment rows (global_prefetch_b8)
      const _Float16* pf =
          A + (size_t)(m0 + (lane & 15) + (lane >> 4) * 16) * lda + k0 + 2 * BK;
      __builtin_prefetch(pf, 0, 1);
    }
#pragma unroll
    for (int j = 0; j < 4; ++j) b[j] = load_frag_b_lds(&Blds[buf][0], nl + 16 * j, lane);

#pragma unroll
    for (int i = 0; i < 4; ++i)
#pragma unroll
      for (int j = 0; j < 4; ++j) acc[i][j] = wmma_f16(a[i], b[j], acc[i][j]);

    wait_async_lds();
    __syncthreads();
  }

#pragma unroll
  for (int i = 0; i < 4; ++i)
#pragma unroll
    for (int j = 0; j < 4; ++j)
      store_frag_c(C, ldc, m0 + 16 * i, nB + nl + 16 * j, acc[i][j], lane);
}

// ---------------------------------------------------------------------------
// Elementwise f32 -> f16 convert
// ---------------------------------------------------------------------------
__global__ void cvt_f32_f16(const float* __restrict__ in, _Float16* __restrict__ out,
                            size_t n) {
  size_t stride = (size_t)gridDim.x * blockDim.x;
  for (size_t i = (size_t)blockIdx.x * blockDim.x + threadIdx.x; i < n; i += stride)
    out[i] = (_Float16)in[i];
}

// ---------------------------------------------------------------------------
// Fused RMSNorm + RoPE over head dim (D=128 threads/block).
// X: (S, NH, D) f32 -> Y: (S, NH, D) f16.  lastOut (if set): f32 row s==S-1.
// ---------------------------------------------------------------------------
__global__ __launch_bounds__(DQ)
void norm_rope(const float* __restrict__ X, const float* __restrict__ w,
               const float* __restrict__ cosT, const float* __restrict__ sinT,
               _Float16* __restrict__ Y, float* __restrict__ lastOut, int NH) {
  const int s = blockIdx.y, hh = blockIdx.x, d = threadIdx.x;
  const float* xp = X + ((size_t)s * NH + hh) * DQ;
  float x = xp[d];

  __shared__ float red[DQ];
  __shared__ float xs[DQ];
  red[d] = x * x;
  __syncthreads();
#pragma unroll
  for (int off = DQ / 2; off > 0; off >>= 1) {
    if (d < off) red[d] += red[d + off];
    __syncthreads();
  }
  float r = rsqrtf(red[0] * (1.0f / DQ) + EPSQ);
  xs[d] = x * r * w[d];
  __syncthreads();

  const int half = DQ / 2;
  float o;
  if (d < half) {
    float c = cosT[(size_t)s * half + d];
    float sn = sinT[(size_t)s * half + d];
    o = xs[d] * c - xs[d + half] * sn;
  } else {
    int dd = d - half;
    float c = cosT[(size_t)s * half + dd];
    float sn = sinT[(size_t)s * half + dd];
    o = xs[dd] * sn + xs[d] * c;
  }
  Y[((size_t)s * NH + hh) * DQ + d] = (_Float16)o;
  if (lastOut && s == SQ - 1) lastOut[hh * DQ + d] = o;
}

// ---------------------------------------------------------------------------
// V pack: Vf (S, KH, D) f32 -> Vt (KH*D, S) f16 (t-contiguous rows for WMMA B),
// plus v_new = V[S-1] in f32.
// ---------------------------------------------------------------------------
__global__ void v_pack(const float* __restrict__ Vf, _Float16* __restrict__ Vt,
                       float* __restrict__ vlast) {
  const int KHD = KHQ * DQ;
  size_t total = (size_t)SQ * KHD;
  size_t stride = (size_t)gridDim.x * blockDim.x;
  for (size_t i = (size_t)blockIdx.x * blockDim.x + threadIdx.x; i < total; i += stride) {
    int t = (int)(i / KHD);
    int r = (int)(i % KHD);          // r = kh*D + d
    float v = Vf[i];
    Vt[(size_t)r * SQ + t] = (_Float16)v;
    if (t == SQ - 1) vlast[r] = v;
  }
}

// ---------------------------------------------------------------------------
// Causal softmax: P[s,t] = softmax_{t<=s}(SC[s,t]*SCALE), zeros for t>s.
// ---------------------------------------------------------------------------
__global__ __launch_bounds__(256)
void softmax_causal(const float* __restrict__ SC, _Float16* __restrict__ P) {
  const int s = blockIdx.x;
  const int len = s + 1;
  const float* row = SC + (size_t)s * SQ;
  _Float16* prow = P + (size_t)s * SQ;
  const int tid = threadIdx.x;
  __shared__ float red[256];

  float m = -3.4e38f;
  for (int t = tid; t < len; t += 256) m = fmaxf(m, row[t] * SCALEQ);
  red[tid] = m; __syncthreads();
#pragma unroll
  for (int off = 128; off > 0; off >>= 1) {
    if (tid < off) red[tid] = fmaxf(red[tid], red[tid + off]);
    __syncthreads();
  }
  m = red[0];
  __syncthreads();

  float sum = 0.0f;
  for (int t = tid; t < len; t += 256) sum += __expf(row[t] * SCALEQ - m);
  red[tid] = sum; __syncthreads();
#pragma unroll
  for (int off = 128; off > 0; off >>= 1) {
    if (tid < off) red[tid] += red[tid + off];
    __syncthreads();
  }
  float inv = 1.0f / red[0];

  for (int t = tid; t < SQ; t += 256) {
    float p = (t < len) ? __expf(row[t] * SCALEQ - m) * inv : 0.0f;
    prow[t] = (_Float16)p;
  }
}

// ---------------------------------------------------------------------------
// Host-side launch
// ---------------------------------------------------------------------------
static inline void launch_gemm(const _Float16* A, const _Float16* B, float* C,
                               int M, int N, int K, int lda, int ldb, int ldc,
                               int mode, hipStream_t st) {
  dim3 blk(32, 4);
  dim3 grd((unsigned)(N / BN), (unsigned)(M / BM));
  gemm_f16f32<<<grd, blk, 0, st>>>(A, B, C, M, N, K, lda, ldb, ldc, mode);
}

extern "C" void kernel_launch(void* const* d_in, const int* in_sizes, int n_in,
                              void* d_out, int out_size, void* d_ws, size_t ws_size,
                              hipStream_t stream) {
  (void)in_sizes; (void)n_in; (void)out_size; (void)ws_size;

  const float* hid   = (const float*)d_in[0];   // (S, HID)
  const float* fcos  = (const float*)d_in[1];   // (S, D/2)
  const float* fsin  = (const float*)d_in[2];   // (S, D/2)
  // d_in[3] = atten_mask : causal mask applied analytically
  const float* Wq    = (const float*)d_in[4];   // (H*D, HID)
  const float* Wk    = (const float*)d_in[5];   // (KH*D, HID)
  const float* Wv    = (const float*)d_in[6];   // (KH*D, HID)
  const float* Wo    = (const float*)d_in[7];   // (HID, H*D)
  const float* qnw   = (const float*)d_in[8];   // (D,)
  const float* knw   = (const float*)d_in[9];   // (D,)

  float* outY = (float*)d_out;                  // (S, HID)
  float* outK = outY + (size_t)SQ * HIDQ;       // (KH, 1, D)
  float* outV = outK + (size_t)KHQ * DQ;        // (KH, 1, D)

  // Workspace layout; regions reused once their producers are dead.
  char* ws = (char*)d_ws;
  const size_t MBy = (size_t)1 << 20;
  _Float16* X16  = (_Float16*)(ws + 0 * MBy);   //  8 MB  (S,HID)      [-> attn16]
  _Float16* Wq16 = (_Float16*)(ws + 8 * MBy);   //  8 MB  (H*D,HID)    [-> AO16]
  _Float16* Wk16 = (_Float16*)(ws + 16 * MBy);  //  4 MB
  _Float16* Wv16 = (_Float16*)(ws + 20 * MBy);  //  4 MB
  _Float16* Wo16 = (_Float16*)(ws + 24 * MBy);  //  8 MB  (HID,H*D)
  float*    Qf   = (float*)   (ws + 32 * MBy);  // 16 MB  (S,H*D)      [-> scores]
  float*    Kf   = (float*)   (ws + 48 * MBy);  //  8 MB  (S,KH*D)     [-> AOf lo]
  float*    Vf   = (float*)   (ws + 56 * MBy);  //  8 MB  (S,KH*D)     [-> AOf hi]
  _Float16* Q16  = (_Float16*)(ws + 64 * MBy);  //  8 MB  (S,H,D)
  _Float16* K16  = (_Float16*)(ws + 72 * MBy);  //  4 MB  (S,KH,D)
  _Float16* Vt16 = (_Float16*)(ws + 76 * MBy);  //  4 MB  (KH,D,S)
  // aliases (producers dead by the time these are written):
  float*    SCf   = Qf;                         // 16 MB  per-head (S,S) scores
  _Float16* AT16  = X16;                        //  8 MB  per-head (S,S) probs
  float*    AOf   = Kf;                         // 16 MB  (S,H*D) attn out f32
  _Float16* AO16  = Wq16;                       //  8 MB  (S,H*D) attn out f16

  // 1) f32 -> f16 conversions
  cvt_f32_f16<<<4096, 256, 0, stream>>>(hid, X16, (size_t)SQ * HIDQ);
  cvt_f32_f16<<<4096, 256, 0, stream>>>(Wq,  Wq16, (size_t)HQ * DQ * HIDQ);
  cvt_f32_f16<<<2048, 256, 0, stream>>>(Wk,  Wk16, (size_t)KHQ * DQ * HIDQ);
  cvt_f32_f16<<<2048, 256, 0, stream>>>(Wv,  Wv16, (size_t)KHQ * DQ * HIDQ);
  cvt_f32_f16<<<4096, 256, 0, stream>>>(Wo,  Wo16, (size_t)HIDQ * HQ * DQ);

  // 2) QKV projections: C = X @ W^T
  launch_gemm(X16, Wq16, Qf, SQ, HQ * DQ,  HIDQ, HIDQ, HIDQ, HQ * DQ,  0, stream);
  launch_gemm(X16, Wk16, Kf, SQ, KHQ * DQ, HIDQ, HIDQ, HIDQ, KHQ * DQ, 0, stream);
  launch_gemm(X16, Wv16, Vf, SQ, KHQ * DQ, HIDQ, HIDQ, HIDQ, KHQ * DQ, 0, stream);

  // 3) RMSNorm + RoPE (K also emits k_new f32); V pack (+ v_new)
  {
    dim3 gq(HQ, SQ), gk(KHQ, SQ);
    norm_rope<<<gq, DQ, 0, stream>>>(Qf, qnw, fcos, fsin, Q16, nullptr, HQ);
    norm_rope<<<gk, DQ, 0, stream>>>(Kf, knw, fcos, fsin, K16, outK, KHQ);
    v_pack<<<2048, 256, 0, stream>>>(Vf, Vt16, outV);
  }

  // 4) Per-head attention
  for (int h = 0; h < HQ; ++h) {
    const int kh = h / GRPQ;
    // scores = Q_h @ K_kh^T  (contraction over d, causal block skip)
    launch_gemm(Q16 + (size_t)h * DQ, K16 + (size_t)kh * DQ, SCf,
                SQ, SQ, DQ, HQ * DQ, KHQ * DQ, SQ, 1, stream);
    // softmax with scale + causal mask, f16 probs
    softmax_causal<<<SQ, 256, 0, stream>>>(SCf, AT16);
    // out_h = P @ V_kh  (contraction over t, causal K-limit)
    launch_gemm(AT16, Vt16 + (size_t)kh * DQ * SQ, AOf + (size_t)h * DQ,
                SQ, DQ, SQ, SQ, SQ, HQ * DQ, 2, stream);
  }

  // 5) Output projection: y = AO @ Wo^T  -> d_out
  cvt_f32_f16<<<4096, 256, 0, stream>>>(AOf, AO16, (size_t)SQ * HQ * DQ);
  launch_gemm(AO16, Wo16, outY, SQ, HIDQ, HQ * DQ, HQ * DQ, HQ * DQ, HIDQ, 0, stream);
}
